// SpatialContextModule_53901839565208
// MI455X (gfx1250) — compile-verified, matrix operand order checked by reading, not run on hardware
//
#include <hip/hip_runtime.h>
#include <hip/hip_bf16.h>
#include <stdint.h>

// ---- types -----------------------------------------------------------------
typedef __bf16 bf16;
typedef __attribute__((ext_vector_type(16))) __bf16 v16bf;
typedef __attribute__((ext_vector_type(8)))  __bf16 bf16x8;
typedef __attribute__((ext_vector_type(8)))  float  v8f;
typedef __attribute__((ext_vector_type(8)))  float  f32x8;

#define BATCH 4
#define CH    256
#define HWDIM 64
#define NPIX  4096   // 64*64
#define NTAPS 36     // 4 dilations * 9 taps
#define NSLAB 8      // row slabs for column-softmax reduction

static __device__ __forceinline__ bf16 f2bf(float f) {
    unsigned u = __builtin_bit_cast(unsigned, f);
    u += 0x7fffu + ((u >> 16) & 1u);               // round-to-nearest-even
    unsigned short h = (unsigned short)(u >> 16);
    return __builtin_bit_cast(bf16, h);
}

static __device__ __forceinline__ v16bf zero_frag() {
    v16bf f;
    bf16 z = __builtin_bit_cast(bf16, (unsigned short)0);
#pragma unroll
    for (int i = 0; i < 16; ++i) f[i] = z;
    return f;
}

// One lane's share of a 16x32 bf16 fragment. rowptr = this lane's matrix
// row (A) / column-row (B); kbase = k0 + lane K-chunk. ISA 7.12.2 layout:
// elems 0..7 -> K=kbase..kbase+7 ; elems 8..15 -> K=kbase+16..kbase+23.
template <typename P>
static __device__ __forceinline__ void frag_load(v16bf& f, P rowptr, int kbase) {
    const bf16x8 lo = *reinterpret_cast<const bf16x8*>(&rowptr[kbase]);
    const bf16x8 hi = *reinterpret_cast<const bf16x8*>(&rowptr[kbase + 16]);
#pragma unroll
    for (int i = 0; i < 8; ++i) { f[i] = lo[i]; f[8 + i] = hi[i]; }
}

static __device__ __forceinline__ v8f wmma_bf16(const v16bf& a, const v16bf& b, const v8f& c) {
    return __builtin_amdgcn_wmma_f32_16x16x32_bf16(false, a, false, b, (short)0, c, false, false);
}

static __device__ __forceinline__ v8f zero_acc() {
    return (v8f){0.f, 0.f, 0.f, 0.f, 0.f, 0.f, 0.f, 0.f};
}

// ---- stage 0: dtype conversion / layout ------------------------------------

// x [b][c][p] f32 (NCHW) -> xbf [b][p][c] bf16
__global__ void cvt_x_kernel(const float* __restrict__ x, bf16* __restrict__ xbf) {
    int idx = blockIdx.x * 256 + threadIdx.x;           // 4,194,304 total
    if (idx >= BATCH * NPIX * CH) return;
    int c = idx & 255;
    int p = (idx >> 8) & 4095;
    int b = idx >> 20;
    xbf[idx] = f2bf(x[((size_t)b * CH + c) * NPIX + p]);
}

// w [d][co][ci][ky][kx] f32 -> wbf [t=d*9+ky*3+kx][co][ci] bf16 (ci contiguous)
__global__ void cvt_w_kernel(const float* __restrict__ w, bf16* __restrict__ wbf) {
    int idx = blockIdx.x * 256 + threadIdx.x;           // 2,359,296 total
    if (idx >= NTAPS * CH * CH) return;
    int ci = idx & 255;
    int co = (idx >> 8) & 255;
    int t  = idx >> 16;
    int d  = t / 9;
    int ky = (t % 9) / 3;
    int kx = t % 3;
    float v = w[((((size_t)d * CH + co) * CH + ci) * 3 + ky) * 3 + kx];
    wbf[idx] = f2bf(v);
}

// ---- stage 1: ASPP as 36 shifted WMMA GEMMs --------------------------------
// Each wave: 32 output pixels x 32 output channels (2x2 register blocking).
__global__ __launch_bounds__(256) void aspp_conv_kernel(
    const bf16* __restrict__ xbf, const bf16* __restrict__ wbf,
    const float* __restrict__ biases,
    bf16* __restrict__ x2bf,       // [b][p][c]
    bf16* __restrict__ x2bfT) {    // [b][c][p]
    const int b    = blockIdx.y;
    const int wave = (blockIdx.x * 256 + threadIdx.x) >> 5;   // 0..1023 per batch
    const int lane = threadIdx.x & 31;
    const int cot  = wave & 7;           // 8 co tiles of 32
    const int pt   = wave >> 3;          // 128 pixel tiles of 32
    const int co0  = cot * 32;
    const int p0   = pt * 32;
    const int ln   = lane & 15;
    const int lk   = lane >> 4;
    const int kb   = lk * 8;

    const int pLo = p0 + ln,      pHi = p0 + 16 + ln;   // A rows (pixels)
    const int pyLo = pLo >> 6, pxLo = pLo & 63;
    const int pyHi = pHi >> 6, pxHi = pHi & 63;

    const bf16* xb = xbf + (size_t)b * NPIX * CH;
    v8f acc[2][2];
#pragma unroll
    for (int tm = 0; tm < 2; ++tm)
#pragma unroll
        for (int tn = 0; tn < 2; ++tn) acc[tm][tn] = zero_acc();

    for (int t = 0; t < NTAPS; ++t) {
        const int d  = (t < 9) ? 6 : (t < 18) ? 12 : (t < 27) ? 18 : 24;
        const int ky = (t % 9) / 3;
        const int kx = t % 3;
        const int dy = d * (ky - 1), dx = d * (kx - 1);
        const int iyLo = pyLo + dy, ixLo = pxLo + dx;
        const int iyHi = pyHi + dy, ixHi = pxHi + dx;
        const bool vLo = (iyLo >= 0) && (iyLo < HWDIM) && (ixLo >= 0) && (ixLo < HWDIM);
        const bool vHi = (iyHi >= 0) && (iyHi < HWDIM) && (ixHi >= 0) && (ixHi < HWDIM);
        const bf16* aRowLo = xb + (size_t)(iyLo * HWDIM + ixLo) * CH;
        const bf16* aRowHi = xb + (size_t)(iyHi * HWDIM + ixHi) * CH;
        const bf16* bRowLo = wbf + ((size_t)t * CH + co0 + ln) * CH;        // B col co0+ln
        const bf16* bRowHi = wbf + ((size_t)t * CH + co0 + 16 + ln) * CH;   // B col co0+16+ln
#pragma unroll
        for (int k0 = 0; k0 < CH; k0 += 32) {
            v16bf a0, a1, b0, b1;
            if (vLo) frag_load(a0, aRowLo, k0 + kb); else a0 = zero_frag();
            if (vHi) frag_load(a1, aRowHi, k0 + kb); else a1 = zero_frag();
            frag_load(b0, bRowLo, k0 + kb);
            frag_load(b1, bRowHi, k0 + kb);
            acc[0][0] = wmma_bf16(a0, b0, acc[0][0]);
            acc[0][1] = wmma_bf16(a0, b1, acc[0][1]);
            acc[1][0] = wmma_bf16(a1, b0, acc[1][0]);
            acc[1][1] = wmma_bf16(a1, b1, acc[1][1]);
        }
    }

    // bias (sum over dilations) + store bf16, row-major and transposed
#pragma unroll
    for (int tm = 0; tm < 2; ++tm) {
#pragma unroll
        for (int tn = 0; tn < 2; ++tn) {
            const int c = co0 + tn * 16 + ln;
            const float bias = biases[c] + biases[CH + c] + biases[2 * CH + c] + biases[3 * CH + c];
            bf16x8 tr;
#pragma unroll
            for (int r = 0; r < 8; ++r) {
                const int p = p0 + tm * 16 + lk * 8 + r;
                const bf16 h = f2bf(acc[tm][tn][r] + bias);
                x2bf[((size_t)b * NPIX + p) * CH + c] = h;
                tr[r] = h;
            }
            *reinterpret_cast<bf16x8*>(
                x2bfT + ((size_t)b * CH + c) * NPIX + p0 + tm * 16 + lk * 8) = tr;
        }
    }
}

// ---- stage 2: fused gram m = x2*x2^T + x*x^T -------------------------------
// Block: 256 i-rows x 32 j-cols; the 32-row j-panels (x2 and x, full K) are
// staged once in LDS and shared by all 8 waves. Each wave: 32x32 output.
__global__ __launch_bounds__(256) void gram_kernel(
    const bf16* __restrict__ xbf, const bf16* __restrict__ x2bf,
    float* __restrict__ m, int b) {
    __shared__ bf16 Bs2[32 * CH];   // x2 rows j0..j0+31, 16 KB
    __shared__ bf16 BsX[32 * CH];   // x  rows j0..j0+31, 16 KB

    const int jt  = blockIdx.x & 127;        // 128 j tiles of 32
    const int itb = blockIdx.x >> 7;         // 16 i blocks of 256
    const int wid = threadIdx.x >> 5;
    const int lane = threadIdx.x & 31;
    const int j0 = jt * 32;
    const int i0 = itb * 256 + wid * 32;
    const int ln = lane & 15;
    const int lk = lane >> 4;
    const int kb = lk * 8;

    const bf16* xb  = xbf  + (size_t)b * NPIX * CH;
    const bf16* x2b = x2bf + (size_t)b * NPIX * CH;

    // cooperative stage of both j-panels: 2 x 16 KB via 128-bit copies
    {
        const uint4* g2 = reinterpret_cast<const uint4*>(x2b + (size_t)j0 * CH);
        const uint4* gx = reinterpret_cast<const uint4*>(xb  + (size_t)j0 * CH);
        uint4* s2 = reinterpret_cast<uint4*>(Bs2);
        uint4* sx = reinterpret_cast<uint4*>(BsX);
#pragma unroll
        for (int u = 0; u < 4; ++u) {
            const int e = threadIdx.x + u * 256;   // 1024 chunks per panel
            s2[e] = g2[e];
            sx[e] = gx[e];
        }
    }
    __syncthreads();

    const bf16* aRow2Lo = x2b + (size_t)(i0 + ln) * CH;
    const bf16* aRow2Hi = x2b + (size_t)(i0 + 16 + ln) * CH;
    const bf16* aRowXLo = xb  + (size_t)(i0 + ln) * CH;
    const bf16* aRowXHi = xb  + (size_t)(i0 + 16 + ln) * CH;
    const bf16* sB2Lo = Bs2 + (size_t)ln * CH;
    const bf16* sB2Hi = Bs2 + (size_t)(16 + ln) * CH;
    const bf16* sBXLo = BsX + (size_t)ln * CH;
    const bf16* sBXHi = BsX + (size_t)(16 + ln) * CH;

    v8f acc[2][2];
#pragma unroll
    for (int tm = 0; tm < 2; ++tm)
#pragma unroll
        for (int tn = 0; tn < 2; ++tn) acc[tm][tn] = zero_acc();

#pragma unroll
    for (int k0 = 0; k0 < CH; k0 += 32) {
        __builtin_prefetch(aRow2Lo + k0 + 64, 0, 0);
        v16bf a0, a1, b0, b1;
        frag_load(a0, aRow2Lo, k0 + kb);
        frag_load(a1, aRow2Hi, k0 + kb);
        frag_load(b0, sB2Lo, k0 + kb);
        frag_load(b1, sB2Hi, k0 + kb);
        acc[0][0] = wmma_bf16(a0, b0, acc[0][0]);
        acc[0][1] = wmma_bf16(a0, b1, acc[0][1]);
        acc[1][0] = wmma_bf16(a1, b0, acc[1][0]);
        acc[1][1] = wmma_bf16(a1, b1, acc[1][1]);
        frag_load(a0, aRowXLo, k0 + kb);
        frag_load(a1, aRowXHi, k0 + kb);
        frag_load(b0, sBXLo, k0 + kb);
        frag_load(b1, sBXHi, k0 + kb);
        acc[0][0] = wmma_bf16(a0, b0, acc[0][0]);
        acc[0][1] = wmma_bf16(a0, b1, acc[0][1]);
        acc[1][0] = wmma_bf16(a1, b0, acc[1][0]);
        acc[1][1] = wmma_bf16(a1, b1, acc[1][1]);
    }

#pragma unroll
    for (int tm = 0; tm < 2; ++tm)
#pragma unroll
        for (int tn = 0; tn < 2; ++tn)
#pragma unroll
            for (int r = 0; r < 8; ++r)
                m[(size_t)(i0 + tm * 16 + lk * 8 + r) * NPIX + j0 + tn * 16 + ln] =
                    acc[tm][tn][r];
}

// ---- stage 3: per-column softmax stats (softmax over axis=1 == rows i) -----
// pass 1: partial max / sum-exp per (slab of 512 rows, column)
__global__ void colstats_partial_kernel(const float* __restrict__ m,
                                        float* __restrict__ pmax,
                                        float* __restrict__ psum) {
    const int j = blockIdx.x * 256 + threadIdx.x;   // 4096 columns
    const int s = blockIdx.y;                        // 8 slabs
    const int ibeg = s * (NPIX / NSLAB);
    float mx = -3.4e38f;
    for (int i = ibeg; i < ibeg + NPIX / NSLAB; ++i)
        mx = fmaxf(mx, m[(size_t)i * NPIX + j]);
    float sum = 0.f;
    for (int i = ibeg; i < ibeg + NPIX / NSLAB; ++i)
        sum += __expf(m[(size_t)i * NPIX + j] - mx);
    pmax[s * NPIX + j] = mx;
    psum[s * NPIX + j] = sum;
}

// pass 2: combine slabs with exp rescaling
__global__ void colstats_combine_kernel(const float* __restrict__ pmax,
                                        const float* __restrict__ psum,
                                        float* __restrict__ cmax,
                                        float* __restrict__ cinv) {
    const int j = blockIdx.x * 256 + threadIdx.x;
    float g = -3.4e38f;
#pragma unroll
    for (int s = 0; s < NSLAB; ++s) g = fmaxf(g, pmax[s * NPIX + j]);
    float sum = 0.f;
#pragma unroll
    for (int s = 0; s < NSLAB; ++s)
        sum += psum[s * NPIX + j] * __expf(pmax[s * NPIX + j] - g);
    cmax[j] = g;
    cinv[j] = 1.0f / sum;
}

// ---- stage 4: x4 = softmax(m) @ x2f, + residual ----------------------------
// Each wave: 32 rows (i) x 64 channels (2x4 register blocking). A fragments
// are exp(m - colmax)*colinv computed on the fly and packed to bf16.
__global__ __launch_bounds__(256) void attn_out_kernel(
    const float* __restrict__ m,
    const float* __restrict__ cmax, const float* __restrict__ cinv,
    const bf16* __restrict__ x2bfT,
    const float* __restrict__ xin, float* __restrict__ out, int b) {
    const int wave = (blockIdx.x * 256 + threadIdx.x) >> 5;   // 512 waves
    const int lane = threadIdx.x & 31;
    const int cg = wave & 3;        // 4 channel groups of 64
    const int it = wave >> 2;       // 128 i tiles of 32
    const int i0 = it * 32;
    const int cbase = cg * 64;
    const int ln = lane & 15;
    const int lk = lane >> 4;
    const int kb = lk * 8;

    const float* mrow0 = m + (size_t)(i0 + ln) * NPIX;
    const float* mrow1 = m + (size_t)(i0 + 16 + ln) * NPIX;
    v8f acc[4][2];
#pragma unroll
    for (int ct = 0; ct < 4; ++ct) { acc[ct][0] = zero_acc(); acc[ct][1] = zero_acc(); }

    for (int j0 = 0; j0 < NPIX; j0 += 32) {
        __builtin_prefetch(mrow0 + j0 + 128, 0, 0);
        const f32x8 mx0 = *reinterpret_cast<const f32x8*>(cmax + j0 + kb);
        const f32x8 mx1 = *reinterpret_cast<const f32x8*>(cmax + j0 + kb + 16);
        const f32x8 iz0 = *reinterpret_cast<const f32x8*>(cinv + j0 + kb);
        const f32x8 iz1 = *reinterpret_cast<const f32x8*>(cinv + j0 + kb + 16);
        const f32x8 u0 = *reinterpret_cast<const f32x8*>(mrow0 + j0 + kb);
        const f32x8 u1 = *reinterpret_cast<const f32x8*>(mrow0 + j0 + kb + 16);
        const f32x8 w0 = *reinterpret_cast<const f32x8*>(mrow1 + j0 + kb);
        const f32x8 w1 = *reinterpret_cast<const f32x8*>(mrow1 + j0 + kb + 16);
        v16bf a0, a1;
#pragma unroll
        for (int i = 0; i < 8; ++i) {
            a0[i]     = f2bf(__expf(u0[i] - mx0[i]) * iz0[i]);
            a0[8 + i] = f2bf(__expf(u1[i] - mx1[i]) * iz1[i]);
            a1[i]     = f2bf(__expf(w0[i] - mx0[i]) * iz0[i]);
            a1[8 + i] = f2bf(__expf(w1[i] - mx1[i]) * iz1[i]);
        }
#pragma unroll
        for (int ct = 0; ct < 4; ++ct) {
            const bf16* brow = x2bfT + ((size_t)b * CH + cbase + ct * 16 + ln) * NPIX;
            v16bf bb;
            frag_load(bb, brow, j0 + kb);
            acc[ct][0] = wmma_bf16(a0, bb, acc[ct][0]);
            acc[ct][1] = wmma_bf16(a1, bb, acc[ct][1]);
        }
    }

    // epilogue: out = x + x4 (NCHW), 8 consecutive pixels per lane per tile
#pragma unroll
    for (int ct = 0; ct < 4; ++ct) {
        const int c = cbase + ct * 16 + ln;
#pragma unroll
        for (int h = 0; h < 2; ++h) {
            const size_t base = ((size_t)b * CH + c) * NPIX + i0 + h * 16 + lk * 8;
#pragma unroll
            for (int r = 0; r < 8; ++r)
                out[base + r] = xin[base + r] + acc[ct][h][r];
        }
    }
}

// ---- host ------------------------------------------------------------------
extern "C" void kernel_launch(void* const* d_in, const int* in_sizes, int n_in,
                              void* d_out, int out_size, void* d_ws, size_t ws_size,
                              hipStream_t stream) {
    const float* x      = (const float*)d_in[0];   // [4,256,64,64]
    const float* w      = (const float*)d_in[1];   // [4,256,256,3,3]
    const float* biases = (const float*)d_in[2];   // [4,256]
    float*       out    = (float*)d_out;           // [4,256,64,64]

    char* ws = (char*)d_ws;
    size_t off = 0;
    auto carve = [&](size_t bytes) -> char* {
        char* p = ws + off;
        off += (bytes + 255) & ~(size_t)255;
        return p;
    };
    bf16*  xbf   = (bf16*)carve((size_t)BATCH * NPIX * CH * sizeof(bf16));   // 8 MB
    bf16*  x2bf  = (bf16*)carve((size_t)BATCH * NPIX * CH * sizeof(bf16));   // 8 MB
    bf16*  x2bfT = (bf16*)carve((size_t)BATCH * CH * NPIX * sizeof(bf16));   // 8 MB
    bf16*  wbf   = (bf16*)carve((size_t)NTAPS * CH * CH * sizeof(bf16));     // 4.5 MB
    float* mbuf  = (float*)carve((size_t)NPIX * NPIX * sizeof(float));       // 64 MB (per batch, reused)
    float* pmax  = (float*)carve((size_t)NSLAB * NPIX * sizeof(float));
    float* psum  = (float*)carve((size_t)NSLAB * NPIX * sizeof(float));
    float* cmax  = (float*)carve((size_t)NPIX * sizeof(float));
    float* cinv  = (float*)carve((size_t)NPIX * sizeof(float));
    if (off > ws_size) return;   // workspace too small; deterministic no-op

    cvt_w_kernel<<<(NTAPS * CH * CH) / 256, 256, 0, stream>>>(w, wbf);
    cvt_x_kernel<<<(BATCH * NPIX * CH) / 256, 256, 0, stream>>>(x, xbf);
    aspp_conv_kernel<<<dim3(128, BATCH), 256, 0, stream>>>(xbf, wbf, biases, x2bf, x2bfT);

    for (int b = 0; b < BATCH; ++b) {
        gram_kernel<<<2048, 256, 0, stream>>>(xbf, x2bf, mbuf, b);
        colstats_partial_kernel<<<dim3(NPIX / 256, NSLAB), 256, 0, stream>>>(mbuf, pmax, psum);
        colstats_combine_kernel<<<NPIX / 256, 256, 0, stream>>>(pmax, psum, cmax, cinv);
        attn_out_kernel<<<64, 256, 0, stream>>>(mbuf, cmax, cinv, x2bfT, x, out, b);
    }
}
